// Collaboration_Module_335007449651
// MI455X (gfx1250) — matrix operand order repaired and checked by reading, not compiled, not run on hardware
//
#include <hip/hip_runtime.h>

// ---------------- problem constants ----------------
#define BROWS 32768
#define C     1000
#define CP    1024          // padded class dim (multiple of 32)
#define NT    (CP/16)       // 64 N-tiles
#define KT    (CP/32)       // 32 K-steps of 32
#define RPB   32            // rows (M) per block: 2 M-tiles of 16
#define NORM  0.031622776601683794f   // 1/sqrt(1000)
#define EPSU  1e-6f

typedef __attribute__((ext_vector_type(16))) __bf16 v16bf;
typedef __attribute__((ext_vector_type(8)))  float  v8f;

__device__ __forceinline__ __bf16 f2bf(float x) { return (__bf16)x; }

// Build an A-fragment (16x16x32 bf16 layout) from an f32 row slice in LDS.
// Lane l: row = l&15, elems 0..7 = K kb..kb+7, elems 8..15 = K kb+16..kb+23,
// where kb = (l>>4)*8 is folded into `ap` by the caller.
__device__ __forceinline__ v16bf load_afrag(const float* ap) {
    float4 f0 = *(const float4*)(ap);
    float4 f1 = *(const float4*)(ap + 4);
    float4 f2 = *(const float4*)(ap + 16);
    float4 f3 = *(const float4*)(ap + 20);
    v16bf a;
    a[0]=f2bf(f0.x);  a[1]=f2bf(f0.y);  a[2]=f2bf(f0.z);  a[3]=f2bf(f0.w);
    a[4]=f2bf(f1.x);  a[5]=f2bf(f1.y);  a[6]=f2bf(f1.z);  a[7]=f2bf(f1.w);
    a[8]=f2bf(f2.x);  a[9]=f2bf(f2.y);  a[10]=f2bf(f2.z); a[11]=f2bf(f2.w);
    a[12]=f2bf(f3.x); a[13]=f2bf(f3.y); a[14]=f2bf(f3.z); a[15]=f2bf(f3.w);
    return a;
}

// ---------------------------------------------------------------------------
// Kernel 1: per-row argmax of p_tar / p_vlm, agreement label (y or -1)
// ---------------------------------------------------------------------------
__global__ __launch_bounds__(256)
void label_kernel(const float* __restrict__ p_tar,
                  const float* __restrict__ p_vlm,
                  int* __restrict__ labels)
{
    __shared__ float bv1[256]; __shared__ int bi1[256];
    __shared__ float bv2[256]; __shared__ int bi2[256];
    const int tid = threadIdx.x;
    const long row = blockIdx.x;

    float v1 = -3.4e38f; int i1 = 0;
    float v2 = -3.4e38f; int i2 = 0;
    for (int n = tid; n < C; n += 256) {
        float a = p_tar[row * C + n];
        if (a > v1) { v1 = a; i1 = n; }
        float b = p_vlm[row * C + n];
        if (b > v2) { v2 = b; i2 = n; }
    }
    bv1[tid] = v1; bi1[tid] = i1; bv2[tid] = v2; bi2[tid] = i2;
    __syncthreads();
    for (int s = 128; s > 0; s >>= 1) {
        if (tid < s) {
            if (bv1[tid+s] > bv1[tid] || (bv1[tid+s] == bv1[tid] && bi1[tid+s] < bi1[tid])) {
                bv1[tid] = bv1[tid+s]; bi1[tid] = bi1[tid+s];
            }
            if (bv2[tid+s] > bv2[tid] || (bv2[tid+s] == bv2[tid] && bi2[tid+s] < bi2[tid])) {
                bv2[tid] = bv2[tid+s]; bi2[tid] = bi2[tid+s];
            }
        }
        __syncthreads();
    }
    if (tid == 0) labels[row] = (bi1[0] == bi2[0]) ? bi1[0] : -1;
}

// ---------------------------------------------------------------------------
// Kernel 2: pack memory_bank into bf16 WMMA B-fragment layout (zero padded).
//   btp: B[k][n] = bank[n][k]  (for S = p_tar · bankᵀ)
//   b2p: B[k][n] = bank[k][n]  (for p_new = atten · bank)
// Fragment layout (16x16x32 bf16): lane l holds column n = l&15,
// element i holds K = (l>>4)*8 + (i&7) + (i>>3)*16. Linear index:
//   ((kt*NT + nt)*32 + lane)*16 + i
// ---------------------------------------------------------------------------
__global__ __launch_bounds__(256)
void pack_kernel(const float* __restrict__ bank,
                 __bf16* __restrict__ btp,
                 __bf16* __restrict__ b2p)
{
    int idx = blockIdx.x * 256 + threadIdx.x;      // total = KT*NT*32*16 = 1,048,576
    const int i    = idx & 15;
    const int lane = (idx >> 4) & 31;
    const int nt   = (idx >> 9) & (NT - 1);
    const int kt   = idx >> 15;
    const int k = kt * 32 + ((lane >> 4) << 3) + (i & 7) + ((i >> 3) << 4);
    const int n = nt * 16 + (lane & 15);
    float v1 = 0.0f, v2 = 0.0f;
    if (k < C && n < C) {
        v1 = bank[(long)n * C + k];   // transposed
        v2 = bank[(long)k * C + n];   // direct
    }
    btp[idx] = f2bf(v1);
    b2p[idx] = f2bf(v2);
}

// ---------------------------------------------------------------------------
// Kernel 3: deterministic per-class masked-mean EMA bank update.
// One block per class; serial scan of labels keeps a fixed accumulation order.
// ---------------------------------------------------------------------------
__global__ __launch_bounds__(256)
void class_update_kernel(const float* __restrict__ p_tar,
                         const int* __restrict__ labels,
                         const float* __restrict__ bank,
                         const float* __restrict__ alpha,
                         float* __restrict__ out_bank)
{
    const int c = blockIdx.x;      // 0..C-1
    const int tid = threadIdx.x;
    float accv[4] = {0.f, 0.f, 0.f, 0.f};
    int cnt = 0;
    for (int b = 0; b < BROWS; ++b) {
        if (labels[b] == c) {
            ++cnt;
            const float* row = p_tar + (long)b * C;
            #pragma unroll
            for (int q = 0; q < 4; ++q) {
                int n = tid + q * 256;
                if (n < C) accv[q] += row[n];
            }
        }
    }
    const float a = alpha[0];
    const float invc = (cnt > 0) ? (1.0f / (float)cnt) : 0.0f;
    #pragma unroll
    for (int q = 0; q < 4; ++q) {
        int n = tid + q * 256;
        if (n < C) {
            float bold = bank[(long)c * C + n];
            out_bank[(long)c * C + n] =
                (cnt > 0) ? (a * bold + (1.0f - a) * (accv[q] * invc)) : bold;
        }
    }
}

// ---------------------------------------------------------------------------
// Kernel 4: fused  GEMM1 -> row softmax -> GEMM2 -> uncertainty mixing.
// Block = 256 threads (8 waves). Block owns 32 rows (2 M-tiles) of p_tar.
// Each wave owns 8 N-tiles; each B fragment is reused for 2 WMMAs (2 M-tiles),
// halving packed-operand L2 traffic vs a 16-row block.
// ---------------------------------------------------------------------------
__global__ __launch_bounds__(256)
void fused_attn_mix_kernel(const float* __restrict__ p_tar,
                           const float* __restrict__ p_vlm,
                           const v16bf* __restrict__ btp,
                           const v16bf* __restrict__ b2p,
                           float* __restrict__ out_mix)
{
    __shared__ float sbuf[RPB * CP];   // 128 KB: p_tar tile, then scores/atten

    const int tid  = threadIdx.x;
    const int lane = tid & 31;
    const int w    = tid >> 5;                 // wave 0..7
    const long rowBase = (long)blockIdx.x * RPB;

    // ---- stage p_tar tile (f32, zero padded to 1024) ----
    for (int e = tid; e < RPB * CP; e += 256) {
        int r = e >> 10, cc = e & (CP - 1);
        sbuf[e] = (cc < C) ? p_tar[(rowBase + r) * C + cc] : 0.0f;
    }
    __syncthreads();

    const int m  = lane & 15;            // A-fragment row within M-tile
    const int kb = (lane >> 4) * 8;      // A-fragment K sub-offset

    v8f acc[2][8];
    #pragma unroll
    for (int mt = 0; mt < 2; ++mt)
        #pragma unroll
        for (int j = 0; j < 8; ++j) { v8f z = {0,0,0,0,0,0,0,0}; acc[mt][j] = z; }

    // ---- GEMM1: S = p_tar · bankᵀ ----
    for (int kt = 0; kt < KT; ++kt) {
        v16bf a0 = load_afrag(&sbuf[(0 * 16 + m) * CP + kt * 32 + kb]);
        v16bf a1 = load_afrag(&sbuf[(1 * 16 + m) * CP + kt * 32 + kb]);
        #pragma unroll
        for (int j = 0; j < 8; ++j) {
            v16bf b = btp[(kt * NT + (w * 8 + j)) * 32 + lane];
            acc[0][j] = __builtin_amdgcn_wmma_f32_16x16x32_bf16(
                false, a0, false, b, (short)0, acc[0][j], false, false);
            acc[1][j] = __builtin_amdgcn_wmma_f32_16x16x32_bf16(
                false, a1, false, b, (short)0, acc[1][j], false, false);
        }
    }
    __syncthreads();   // done reading p_tar tile

    // ---- scatter scaled scores into LDS (C/D layout: M = r + 8*(l>>4), N = l&15)
    #pragma unroll
    for (int mt = 0; mt < 2; ++mt) {
        #pragma unroll
        for (int j = 0; j < 8; ++j) {
            int cc = w * 128 + j * 16 + (lane & 15);
            #pragma unroll
            for (int r = 0; r < 8; ++r) {
                int mm = mt * 16 + r + 8 * (lane >> 4);
                sbuf[mm * CP + cc] = acc[mt][j][r] * NORM;
            }
        }
    }
    __syncthreads();

    // ---- row softmax over 1000 valid columns (scratch lives in pad cols) ----
    {
        const int row = tid >> 3, sub = tid & 7;   // 32 rows x 8 threads
        float mx = -3.4e38f;
        for (int n = sub; n < C; n += 8) mx = fmaxf(mx, sbuf[row * CP + n]);
        sbuf[row * CP + C + sub] = mx;
        __syncthreads();
        float rmx = -3.4e38f;
        #pragma unroll
        for (int s = 0; s < 8; ++s) rmx = fmaxf(rmx, sbuf[row * CP + C + s]);
        __syncthreads();
        float ps = 0.0f;
        for (int n = sub; n < C; n += 8) {
            float e = __expf(sbuf[row * CP + n] - rmx);
            sbuf[row * CP + n] = e;
            ps += e;
        }
        sbuf[row * CP + C + sub] = ps;
        __syncthreads();
        float rs = 0.0f;
        #pragma unroll
        for (int s = 0; s < 8; ++s) rs += sbuf[row * CP + C + s];
        const float inv = 1.0f / rs;
        __syncthreads();
        for (int n = sub; n < C; n += 8) sbuf[row * CP + n] *= inv;
        for (int n = C + sub; n < CP; n += 8) sbuf[row * CP + n] = 0.0f;  // clear pad
    }
    __syncthreads();

    // ---- GEMM2: p_new = atten · bank ----
    #pragma unroll
    for (int mt = 0; mt < 2; ++mt)
        #pragma unroll
        for (int j = 0; j < 8; ++j) { v8f z = {0,0,0,0,0,0,0,0}; acc[mt][j] = z; }
    for (int kt = 0; kt < KT; ++kt) {
        v16bf a0 = load_afrag(&sbuf[(0 * 16 + m) * CP + kt * 32 + kb]);
        v16bf a1 = load_afrag(&sbuf[(1 * 16 + m) * CP + kt * 32 + kb]);
        #pragma unroll
        for (int j = 0; j < 8; ++j) {
            v16bf b = b2p[(kt * NT + (w * 8 + j)) * 32 + lane];
            acc[0][j] = __builtin_amdgcn_wmma_f32_16x16x32_bf16(
                false, a0, false, b, (short)0, acc[0][j], false, false);
            acc[1][j] = __builtin_amdgcn_wmma_f32_16x16x32_bf16(
                false, a1, false, b, (short)0, acc[1][j], false, false);
        }
    }

    // ---- fused uncertainty mixing epilogue ----
    #pragma unroll
    for (int mt = 0; mt < 2; ++mt) {
        #pragma unroll
        for (int j = 0; j < 8; ++j) {
            int cc = w * 128 + j * 16 + (lane & 15);
            if (cc < C) {
                #pragma unroll
                for (int r = 0; r < 8; ++r) {
                    int mm = mt * 16 + r + 8 * (lane >> 4);
                    long grow = rowBase + mm;
                    float t  = acc[mt][j][r];
                    float pv = p_vlm[grow * C + cc];
                    float eut = __expf(t  * __logf(t  + EPSU));  // exp(-u(t))
                    float euv = __expf(pv * __logf(pv + EPSU));  // exp(-u(pv))
                    out_mix[grow * C + cc] = (eut * t + euv * pv) / (eut + euv);
                }
            }
        }
    }
}

// ---------------------------------------------------------------------------
// launch
// ---------------------------------------------------------------------------
extern "C" void kernel_launch(void* const* d_in, const int* in_sizes, int n_in,
                              void* d_out, int out_size, void* d_ws, size_t ws_size,
                              hipStream_t stream) {
    const float* p_tar = (const float*)d_in[0];
    const float* p_vlm = (const float*)d_in[1];
    const float* alpha = (const float*)d_in[2];
    const float* bank  = (const float*)d_in[3];

    float* out_mix  = (float*)d_out;                       // [B, C]
    float* out_bank = out_mix + (long)BROWS * C;           // [C, C]

    char* ws = (char*)d_ws;
    // workspace layout (all offsets 32B-aligned)
    int*    labels = (int*)ws;                             // 32768 * 4 = 131072 B
    __bf16* btp    = (__bf16*)(ws + 131072);               // 1,048,576 * 2 = 2 MB (4 MB reserved)
    __bf16* b2p    = (__bf16*)(ws + 131072 + 4194304);     // 1,048,576 * 2 = 2 MB

    // 1. agreement labels
    label_kernel<<<BROWS, 256, 0, stream>>>(p_tar, p_vlm, labels);
    // 2. pack bank into bf16 WMMA fragments (both orientations)
    pack_kernel<<<(KT * NT * 32 * 16) / 256, 256, 0, stream>>>(bank, btp, b2p);
    // 3. deterministic per-class EMA bank update
    class_update_kernel<<<C, 256, 0, stream>>>(p_tar, labels, bank, alpha, out_bank);
    // 4. fused attention + softmax + mixing (WMMA path, 32 rows/block)
    fused_attn_mix_kernel<<<BROWS / RPB, 256, 0, stream>>>(
        p_tar, p_vlm, (const v16bf*)btp, (const v16bf*)b2p, out_mix);
}
